// DualGCN_36636071035172
// MI455X (gfx1250) — matrix operand order repaired and checked by reading, not compiled, not run on hardware
//
#include <hip/hip_runtime.h>
#include <hip/hip_bf16.h>

typedef __attribute__((ext_vector_type(16))) _Float16 v16h;
typedef __attribute__((ext_vector_type(8)))  _Float16 v8h;
typedef __attribute__((ext_vector_type(8)))  float    v8f;

// ---------------------------------------------------------------------------
// Small elementwise kernels
// ---------------------------------------------------------------------------
__global__ void fill_k(float* p, int n, float v) {
    int i = blockIdx.x * blockDim.x + threadIdx.x;
    if (i < n) p[i] = v;
}

__global__ void deg_acc_k(float* deg, const int* __restrict__ cols,
                          const float* __restrict__ w, int E) {
    int i = blockIdx.x * blockDim.x + threadIdx.x;
    if (i < E) unsafeAtomicAdd(&deg[cols[i]], w[i]);
}

__global__ void dinv_fin_k(float* d, int n) {
    int i = blockIdx.x * blockDim.x + threadIdx.x;
    if (i < n) {
        float v = d[i];
        d[i] = (v > 0.f) ? __frsqrt_rn(v) : 0.f;
    }
}

__global__ void relu_k(float* x, long long n) {
    long long i = (long long)blockIdx.x * blockDim.x + threadIdx.x;
    if (i < n) x[i] = fmaxf(x[i], 0.f);
}

// out[i,f] = dinv[i]^2 * h[i,f] + bias[f]   (self-loop term + conv bias)
__global__ void init_agg_k(float* __restrict__ out, const float* __restrict__ h,
                           const float* __restrict__ bias, const float* __restrict__ dinv,
                           int N, int H) {
    long long tid = (long long)blockIdx.x * blockDim.x + threadIdx.x;
    long long total = (long long)N * H;
    if (tid >= total) return;
    int i = (int)(tid / H);
    int f = (int)(tid - (long long)i * H);
    float di = dinv[i];
    out[tid] = di * di * h[tid] + bias[f];
}

// out[col] += dinv[row]*w*dinv[col] * h[row]   (float4 per thread, HW f32 atomics)
__global__ void aggr_k(const float* __restrict__ h, const int* __restrict__ rows,
                       const int* __restrict__ cols, const float* __restrict__ w,
                       const float* __restrict__ dinv, float* __restrict__ out,
                       long long E, int lg /* H = 4<<lg */) {
    long long tid = (long long)blockIdx.x * blockDim.x + threadIdx.x;
    long long total = E << lg;
    if (tid >= total) return;
    long long e = tid >> lg;
    int c = (int)(tid & ((1 << lg) - 1));
    int r = rows[e];
    int cc = cols[e];
    float nw = dinv[r] * w[e] * dinv[cc];
    int H = 4 << lg;
    const float4 hv = *(const float4*)(h + (size_t)r * H + (c << 2));
    float* op = out + (size_t)cc * H + (c << 2);
    unsafeAtomicAdd(op + 0, nw * hv.x);
    unsafeAtomicAdd(op + 1, nw * hv.y);
    unsafeAtomicAdd(op + 2, nw * hv.z);
    unsafeAtomicAdd(op + 3, nw * hv.w);
}

__global__ void concat_k(const float* __restrict__ a, const float* __restrict__ b,
                         float* __restrict__ o, int N, int H1, int H2) {
    long long tid = (long long)blockIdx.x * blockDim.x + threadIdx.x;
    int HT = H1 + H2;
    long long total = (long long)N * HT;
    if (tid >= total) return;
    int i = (int)(tid / HT);
    int j = (int)(tid - (long long)i * HT);
    o[tid] = (j < H1) ? a[(size_t)i * H1 + j] : b[(size_t)i * H2 + (j - H1)];
}

// ---------------------------------------------------------------------------
// WMMA GEMM: C[M,N] = A[M,K] @ W[K,N] (+ bias), f32 in, f16 WMMA, f32 acc.
// Block = 256 threads (8 waves). Tile 64(M) x 64(N), K stepped by 32.
// Wave w: wm = w%4 -> M sub 16*wm, wn = w/4 -> N strip 32*wn (two 16x16 frags).
// Requires: N % 64 == 0, K % 32 == 0 (all shapes here satisfy this).
// ---------------------------------------------------------------------------
__global__ __launch_bounds__(256)
void gemm_wmma_k(const float* __restrict__ A, const float* __restrict__ W,
                 const float* __restrict__ bias, float* __restrict__ C,
                 int M, int K, int N) {
    // rows padded to 40 halfs (80B): keeps 16B alignment of 8-half chunks
    __shared__ __attribute__((aligned(16))) _Float16 As[64][40]; // [m][k]
    __shared__ __attribute__((aligned(16))) _Float16 Bs[64][40]; // [n][k] (transposed)

    const int t    = threadIdx.x;
    const int wv   = t >> 5;
    const int lane = t & 31;
    const int wm   = wv & 3;
    const int wn   = wv >> 2;
    const int half = lane >> 4;   // 0: lanes 0-15, 1: lanes 16-31
    const int lm   = lane & 15;

    const int m0 = blockIdx.x * 64;
    const int n0 = blockIdx.y * 64;

    const int lr = t >> 2;        // 0..63 : A row / B col within tile
    const int lc = (t & 3) * 8;   // 0,8,16,24 : k sub-chunk

    v8f c0 = {};
    v8f c1 = {};

    for (int k0 = 0; k0 < K; k0 += 32) {
        __syncthreads();
        // --- stage A tile (f32 -> f16) ---
        {
            int row = m0 + lr;
            if (row < M) {
                const float* ap = A + (size_t)row * K + k0 + lc;
                #pragma unroll
                for (int j = 0; j < 8; ++j) As[lr][lc + j] = (_Float16)ap[j];
            } else {
                #pragma unroll
                for (int j = 0; j < 8; ++j) As[lr][lc + j] = (_Float16)0.f;
            }
        }
        // --- stage B tile transposed: Bs[n][k] = W[(k0+k)*N + n0+n] ---
        {
            const float* wp = W + (size_t)(k0 + lc) * N + n0 + lr;
            #pragma unroll
            for (int j = 0; j < 8; ++j) Bs[lr][lc + j] = (_Float16)wp[(size_t)j * N];
        }
        __syncthreads();

        // --- A fragment (contiguous k order, same as B) ---
        // lanes 0-15: K 0..15; lanes 16-31: K 16..31
        const int arow = wm * 16 + lm;
        const int koff = half * 16;
        v8h a0 = *(const v8h*)&As[arow][koff];
        v8h a1 = *(const v8h*)&As[arow][koff + 8];
        v16h a;
        #pragma unroll
        for (int j = 0; j < 8; ++j) { a[j] = a0[j]; a[8 + j] = a1[j]; }

        // --- B fragments (lanes 0-15: K 0..15; lanes 16-31: K 16..31) ---
        const int kb  = half * 16;
        const int nb0 = wn * 32 + lm;
        v8h b00 = *(const v8h*)&Bs[nb0][kb];
        v8h b01 = *(const v8h*)&Bs[nb0][kb + 8];
        v16h b0;
        #pragma unroll
        for (int j = 0; j < 8; ++j) { b0[j] = b00[j]; b0[8 + j] = b01[j]; }

        const int nb1 = nb0 + 16;
        v8h b10 = *(const v8h*)&Bs[nb1][kb];
        v8h b11 = *(const v8h*)&Bs[nb1][kb + 8];
        v16h b1;
        #pragma unroll
        for (int j = 0; j < 8; ++j) { b1[j] = b10[j]; b1[8 + j] = b11[j]; }

        c0 = __builtin_amdgcn_wmma_f32_16x16x32_f16(false, a, false, b0, (short)0, c0, false, false);
        c1 = __builtin_amdgcn_wmma_f32_16x16x32_f16(false, a, false, b1, (short)0, c1, false, false);
    }

    // --- store C (f32 16x16 C/D layout: VGPR r -> M = r + 8*half, N = lm) ---
    const int col0 = n0 + wn * 32 + lm;
    #pragma unroll
    for (int r = 0; r < 8; ++r) {
        int row = m0 + wm * 16 + half * 8 + r;
        if (row >= M) continue;
        float v0 = c0[r];
        float v1 = c1[r];
        if (bias) { v0 += bias[col0]; v1 += bias[col0 + 16]; }
        C[(size_t)row * N + col0]      = v0;
        C[(size_t)row * N + col0 + 16] = v1;
    }
}

// ---------------------------------------------------------------------------
// Host launcher
// ---------------------------------------------------------------------------
static inline unsigned gb(long long n) { return (unsigned)((n + 255) / 256); }

extern "C" void kernel_launch(void* const* d_in, const int* in_sizes, int n_in,
                              void* d_out, int out_size, void* d_ws, size_t ws_size,
                              hipStream_t stream) {
    const float* xRNA  = (const float*)d_in[0];
    const float* xADT  = (const float*)d_in[1];
    const int*   simI  = (const int*)d_in[2];
    const float* simW  = (const float*)d_in[3];
    const int*   distI = (const int*)d_in[4];
    const float* distW = (const float*)d_in[5];
    const int*   comI  = (const int*)d_in[6];
    const float* comW  = (const float*)d_in[7];
    const float* W1  = (const float*)d_in[8];
    const float* b1  = (const float*)d_in[9];
    const float* W2  = (const float*)d_in[10];
    const float* b2  = (const float*)d_in[11];
    const float* Wp  = (const float*)d_in[12];
    const float* bp  = (const float*)d_in[13];
    const float* Ws  = (const float*)d_in[14];
    const float* bs  = (const float*)d_in[15];
    const float* Wd  = (const float*)d_in[16];
    const float* bd  = (const float*)d_in[17];
    const float* Wf1 = (const float*)d_in[18];
    const float* bf1 = (const float*)d_in[19];
    const float* Wf2 = (const float*)d_in[20];
    const float* bf2 = (const float*)d_in[21];

    const int N  = in_sizes[0] / 256;   // IN = 256
    const int Es = in_sizes[3];
    const int Ed = in_sizes[5];
    const int Ec = in_sizes[7];
    const int HID = 512, OUT = 128;

    float* ws = (float*)d_ws;
    size_t o = 0;
    float* dinvS = ws + o; o += N;
    float* dinvD = ws + o; o += N;
    float* dinvC = ws + o; o += N;
    float* bufA  = ws + o; o += (size_t)N * HID;   // [N,512]
    float* bufB  = ws + o; o += (size_t)N * HID;   // [N,512]
    float* bufC  = ws + o; o += (size_t)N * OUT;   // [N,128]

    float* out    = (float*)d_out;
    float* x_sim  = out;
    float* x_dist = out + (size_t)N * OUT;
    float* fused  = out + 2 * (size_t)N * OUT;
    float* fusedp = out + 3 * (size_t)N * OUT;
    float* pro    = out + 4 * (size_t)N * OUT;

    const dim3 blk(256);
    const dim3 g512((N + 63) / 64, HID / 64);
    const dim3 g128((N + 63) / 64, OUT / 64);

    // --- symmetric-norm degree factors (deg init 1.0 = self-loop weight) ---
    fill_k<<<gb(N), blk, 0, stream>>>(dinvS, N, 1.0f);
    fill_k<<<gb(N), blk, 0, stream>>>(dinvD, N, 1.0f);
    fill_k<<<gb(N), blk, 0, stream>>>(dinvC, N, 1.0f);
    deg_acc_k<<<gb(Es), blk, 0, stream>>>(dinvS, simI + Es, simW, Es);
    deg_acc_k<<<gb(Ed), blk, 0, stream>>>(dinvD, distI + Ed, distW, Ed);
    deg_acc_k<<<gb(Ec), blk, 0, stream>>>(dinvC, comI + Ec, comW, Ec);
    dinv_fin_k<<<gb(N), blk, 0, stream>>>(dinvS, N);
    dinv_fin_k<<<gb(N), blk, 0, stream>>>(dinvD, N);
    dinv_fin_k<<<gb(N), blk, 0, stream>>>(dinvC, N);

    // --- sim branch: xs = relu(agg_sim(xRNA@W1) + b1); x_sim = agg_sim(xs@Ws) + bs
    gemm_wmma_k<<<g512, blk, 0, stream>>>(xRNA, W1, nullptr, bufA, N, 256, HID);
    init_agg_k<<<gb((long long)N * HID), blk, 0, stream>>>(bufB, bufA, b1, dinvS, N, HID);
    aggr_k<<<gb((long long)Es << 7), blk, 0, stream>>>(bufA, simI, simI + Es, simW, dinvS, bufB, Es, 7);
    relu_k<<<gb((long long)N * HID), blk, 0, stream>>>(bufB, (long long)N * HID);
    gemm_wmma_k<<<g128, blk, 0, stream>>>(bufB, Ws, nullptr, bufC, N, HID, OUT);
    init_agg_k<<<gb((long long)N * OUT), blk, 0, stream>>>(x_sim, bufC, bs, dinvS, N, OUT);
    aggr_k<<<gb((long long)Es << 5), blk, 0, stream>>>(bufC, simI, simI + Es, simW, dinvS, x_sim, Es, 5);

    // --- dist branch ---
    gemm_wmma_k<<<g512, blk, 0, stream>>>(xRNA, W2, nullptr, bufA, N, 256, HID);
    init_agg_k<<<gb((long long)N * HID), blk, 0, stream>>>(bufB, bufA, b2, dinvD, N, HID);
    aggr_k<<<gb((long long)Ed << 7), blk, 0, stream>>>(bufA, distI, distI + Ed, distW, dinvD, bufB, Ed, 7);
    relu_k<<<gb((long long)N * HID), blk, 0, stream>>>(bufB, (long long)N * HID);
    gemm_wmma_k<<<g128, blk, 0, stream>>>(bufB, Wd, nullptr, bufC, N, HID, OUT);
    init_agg_k<<<gb((long long)N * OUT), blk, 0, stream>>>(x_dist, bufC, bd, dinvD, N, OUT);
    aggr_k<<<gb((long long)Ed << 5), blk, 0, stream>>>(bufC, distI, distI + Ed, distW, dinvD, x_dist, Ed, 5);

    // --- protein branch: pro = agg_com(xADT@Wp) + bp ---
    gemm_wmma_k<<<g128, blk, 0, stream>>>(xADT, Wp, nullptr, bufC, N, 64, OUT);
    init_agg_k<<<gb((long long)N * OUT), blk, 0, stream>>>(pro, bufC, bp, dinvC, N, OUT);
    aggr_k<<<gb((long long)Ec << 5), blk, 0, stream>>>(bufC, comI, comI + Ec, comW, dinvC, pro, Ec, 5);

    // --- fusion: fused = [x_sim|x_dist]@Wf1 + bf1 ; fused_pro = [fused|pro]@Wf2 + bf2
    concat_k<<<gb((long long)N * 256), blk, 0, stream>>>(x_sim, x_dist, bufA, N, OUT, OUT);
    gemm_wmma_k<<<g128, blk, 0, stream>>>(bufA, Wf1, bf1, fused, N, 256, OUT);
    concat_k<<<gb((long long)N * 256), blk, 0, stream>>>(fused, pro, bufB, N, OUT, OUT);
    gemm_wmma_k<<<g128, blk, 0, stream>>>(bufB, Wf2, bf2, fusedp, N, 256, OUT);
}
